// ResNet_1735166787587
// MI455X (gfx1250) — compile-verified, matrix operand order checked by reading, not run on hardware
//
#include <hip/hip_runtime.h>
#include <stdint.h>

// ---------------- types for WMMA (CDNA5 / gfx1250, wave32) ----------------
typedef __attribute__((ext_vector_type(16))) __bf16 v16bf;
typedef __attribute__((ext_vector_type(8)))  float  v8f;
typedef __attribute__((ext_vector_type(4)))  int    i4v;

union V16U { uint4 u[2]; v16bf v; };

__device__ __forceinline__ unsigned short f2bf(float f) {
  union { float f; unsigned int u; } cv; cv.f = f;
  unsigned int u = cv.u;
  u += 0x7FFFu + ((u >> 16) & 1u);   // round-to-nearest-even
  return (unsigned short)(u >> 16);
}

// ---------------- CDNA5 async global->LDS copy (ASYNCcnt-tracked) ----------
#if __has_builtin(__builtin_amdgcn_global_load_async_to_lds_b128)
#define HAVE_ASYNC 1
__device__ __forceinline__ void copyB16(const unsigned short* g, unsigned short* l) {
  __builtin_amdgcn_global_load_async_to_lds_b128(
      (__attribute__((address_space(1))) i4v*)g,
      (__attribute__((address_space(3))) i4v*)l, 0, 0);
}
#else
#define HAVE_ASYNC 0
__device__ __forceinline__ void copyB16(const unsigned short* g, unsigned short* l) {
  *reinterpret_cast<uint4*>(l) = *reinterpret_cast<const uint4*>(g);
}
#endif

#if __has_builtin(__builtin_amdgcn_s_wait_asynccnt)
#define WAIT_ASYNC(n) __builtin_amdgcn_s_wait_asynccnt(n)
#elif HAVE_ASYNC
#define WAIT_ASYNC(n) asm volatile("s_wait_asynccnt %0" :: "n"(n) : "memory")
#else
#define WAIT_ASYNC(n) ((void)0)
#endif

// ---------------------------------------------------------------------------
// Kernel 1: weight transpose + bf16 convert: W[k][cin][cout] f32 -> Wt[k][cout][cin] bf16
// ---------------------------------------------------------------------------
__global__ void wt_kernel(const float* __restrict__ W, unsigned short* __restrict__ Wt,
                          int Cin, int Cout, int total) {
  int t = blockIdx.x * blockDim.x + threadIdx.x;
  if (t >= total) return;
  int per = Cin * Cout;
  int k = t / per;
  int r = t - k * per;
  int i = r / Cout;
  int o = r - i * Cout;
  Wt[((size_t)k * Cout + o) * Cin + i] = f2bf(W[t]);
}

// ---------------------------------------------------------------------------
// Kernel 2: stem conv (Cin=1): out[i][c] = sum_e x[j_e] * W[k_e][c]
// ---------------------------------------------------------------------------
__global__ void stem_kernel(const float* __restrict__ x, const int* __restrict__ trip,
                            const float* __restrict__ W, float* __restrict__ out) {
  int t = blockIdx.x * blockDim.x + threadIdx.x;    // 32768*64 threads
  int i = t >> 6, c = t & 63;
  const int E = 32768 * 32;
  float acc = 0.f;
  for (int e = 0; e < 32; ++e) {
    int j = trip[E + i * 32 + e];
    int k = trip[2 * E + i * 32 + e];
    acc = fmaf(x[j], W[k * 64 + c], acc);
  }
  out[t] = acc;
}

// ---------------------------------------------------------------------------
// Kernel 3: ragged layer norm (per-sample stats) + optional residual + relu.
// ---------------------------------------------------------------------------
__global__ void rln_kernel(const float* __restrict__ x, const float* __restrict__ g,
                           const float* __restrict__ b, const float* __restrict__ res,
                           float* __restrict__ out, int n_per, int C, int relu) {
  const int smp = blockIdx.x;
  const int total = n_per * C;
  const size_t base = (size_t)smp * total;
  __shared__ float s1[1024], s2[1024];
  float sum = 0.f, sq = 0.f;
  for (int t = threadIdx.x; t < total; t += blockDim.x) {
    float v = x[base + t]; sum += v; sq = fmaf(v, v, sq);
  }
  s1[threadIdx.x] = sum; s2[threadIdx.x] = sq;
  __syncthreads();
  for (int s = blockDim.x >> 1; s > 0; s >>= 1) {
    if ((int)threadIdx.x < s) { s1[threadIdx.x] += s1[threadIdx.x + s]; s2[threadIdx.x] += s2[threadIdx.x + s]; }
    __syncthreads();
  }
  float inv = 1.0f / (float)total;
  float mu = s1[0] * inv;
  float var = s2[0] * inv - mu * mu;
  float rstd = rsqrtf(var + 1e-5f);
  for (int t = threadIdx.x; t < total; t += blockDim.x) {
    int c = t % C;
    float y = (x[base + t] - mu) * rstd * g[c] + b[c];
    if (res) y += res[base + t];
    if (relu) y = fmaxf(y, 0.f);
    out[base + t] = y;
  }
}

// ---------------------------------------------------------------------------
// Kernel 4: max-pool over 8 gathered rows
// ---------------------------------------------------------------------------
__global__ void pool_kernel(const float* __restrict__ x, const int* __restrict__ pidx,
                            int n_out, int C, float* __restrict__ out) {
  int t = blockIdx.x * blockDim.x + threadIdx.x;
  if (t >= n_out * C) return;
  int r = t / C, c = t - r * C;
  float m = -3.4e38f;
  for (int p = 0; p < 8; ++p)
    m = fmaxf(m, x[(size_t)pidx[r * 8 + p] * C + c]);
  out[t] = m;
}

// ---------------------------------------------------------------------------
// Kernel 5: scatter/segment-sum, LDS-staged per output point, bf16 output.
// ---------------------------------------------------------------------------
__global__ void scatter_kernel(const float* __restrict__ x, const int* __restrict__ trip,
                               int n_out, int Cin, unsigned short* __restrict__ A) {
  extern __shared__ float lds[];                 // 27*Cin floats
  const int i = blockIdx.x;
  const int E = n_out * 16;
  const int tot = 27 * Cin;
  if (threadIdx.x < 16)
    __builtin_prefetch(x + (size_t)trip[E + i * 16 + threadIdx.x] * Cin, 0, 0);
  for (int t = threadIdx.x; t < tot; t += blockDim.x) lds[t] = 0.f;
  __syncthreads();
  for (int e = 0; e < 16; ++e) {
    int j = trip[E + i * 16 + e];
    int k = trip[2 * E + i * 16 + e];
    const float* xr = x + (size_t)j * Cin;
    for (int c = threadIdx.x; c < Cin; c += blockDim.x)
      lds[k * Cin + c] += xr[c];                 // channel c owned by one thread: race-free
  }
  __syncthreads();
  for (int t = threadIdx.x; t < tot; t += blockDim.x) {
    int k = t / Cin, c = t - k * Cin;
    A[((size_t)k * n_out + i) * Cin + c] = f2bf(lds[t]);
  }
}

// ---------------------------------------------------------------------------
// Kernel 6: row gather -> bf16 (for 1x1 downsample conv input)
// ---------------------------------------------------------------------------
__global__ void gather_bf16_kernel(const float* __restrict__ x, const int* __restrict__ idx,
                                   int n, int C, unsigned short* __restrict__ out) {
  int t = blockIdx.x * blockDim.x + threadIdx.x;
  if (t >= n * C) return;
  int r = t / C, c = t - r * C;
  out[t] = f2bf(x[(size_t)idx[r] * C + c]);
}

// ---------------------------------------------------------------------------
// Kernel 7: batched WMMA GEMM: out[M][N] = sum_{k<K3} A_k[M][Cin] @ W_k[Cin][N]
// Block = 8 waves = 128(M)x64(N) tile. B tile (64x64K bf16, 8KB) staged in LDS
// via async global->LDS copies, double buffered; A operand ping-pong in
// registers with 2x-unrolled loop (no inter-step register copies -> no
// WMMA->VALU hazard NOPs). K-step = 64 (8 WMMAs per step per wave).
// ---------------------------------------------------------------------------
#define BCOL 72   // shorts per column segment in LDS (64 data + 8 pad)

__global__ __launch_bounds__(256)
void wmma_gemm_kernel(const unsigned short* __restrict__ A,
                      const unsigned short* __restrict__ Wt,
                      float* __restrict__ out,
                      int M, int N, int Cin, int K3) {
  __shared__ unsigned short ldsB[2][64 * BCOL];
  const int tid = threadIdx.x;
  const int lane = tid & 31, wv = tid >> 5;
  const int m = lane & 15, h = lane >> 4;
  const int nTiles = N >> 6;
  const int tileM = (int)blockIdx.x / nTiles;
  const int tileN = (int)blockIdx.x - tileM * nTiles;
  const int colBase = tileN * 64;
  const int rowA = tileM * 128 + wv * 16 + m;

  // B copy mapping: 512 16B-chunks (64 cols x 8), 2 per thread
  const int c0col = tid >> 3;
  const int c1col = 32 + (tid >> 3);
  const int cch   = tid & 7;

  const int kcSteps = Cin >> 6;                       // power of two: 1,2,4,8
  const int kcShift = (kcSteps == 1) ? 0 : (kcSteps == 2 ? 1 : (kcSteps == 4 ? 2 : 3));
  const int kcMask  = kcSteps - 1;
  const int S = K3 * kcSteps;

  v8f acc[4];
  #pragma unroll
  for (int nt = 0; nt < 4; ++nt) acc[nt] = (v8f){0.f,0.f,0.f,0.f,0.f,0.f,0.f,0.f};

  auto issueB = [&](int k3, int kc, int buf) {
    const unsigned short* g0 = Wt + ((size_t)k3 * N + colBase + c0col) * Cin + kc + cch * 8;
    const unsigned short* g1 = Wt + ((size_t)k3 * N + colBase + c1col) * Cin + kc + cch * 8;
    unsigned short* lb = &ldsB[buf][0];
    copyB16(g0, lb + c0col * BCOL + cch * 8);
    copyB16(g1, lb + c1col * BCOL + cch * 8);
  };
  auto loadA = [&](int k3, int kc, V16U& x0, V16U& x1) {
    const unsigned short* ap = A + ((size_t)k3 * M + rowA) * Cin + kc + h * 8;
    x0.u[0] = *(const uint4*)ap;        x0.u[1] = *(const uint4*)(ap + 16);
    x1.u[0] = *(const uint4*)(ap + 32); x1.u[1] = *(const uint4*)(ap + 48);
  };
  // one pipeline step: consume (a0c,a1c) from ldsB[buf], prefetch step s+1 into
  // ldsB[buf^1] and (a0n,a1n)
  auto step = [&](int s, int buf, V16U& a0c, V16U& a1c, V16U& a0n, V16U& a1n) {
    int sn = s + 1;
    if (sn < S) {
      int k3n = sn >> kcShift, kcn = (sn & kcMask) << 6;
      issueB(k3n, kcn, buf ^ 1);
      loadA(k3n, kcn, a0n, a1n);
      WAIT_ASYNC(2);                     // copies for ldsB[buf] (issued last step) done
    } else {
      WAIT_ASYNC(0);
    }
    __syncthreads();                     // ldsB[buf] visible to all waves
    const unsigned short* lb = &ldsB[buf][0];
    #pragma unroll
    for (int nt = 0; nt < 4; ++nt) {
      const unsigned short* seg = lb + (nt * 16 + m) * BCOL;
      V16U b0, b1;
      b0.u[0] = *(const uint4*)(seg + h * 8);       b0.u[1] = *(const uint4*)(seg + h * 8 + 16);
      b1.u[0] = *(const uint4*)(seg + 32 + h * 8);  b1.u[1] = *(const uint4*)(seg + 32 + h * 8 + 16);
      acc[nt] = __builtin_amdgcn_wmma_f32_16x16x32_bf16(false, a0c.v, false, b0.v,
                                                        (short)0, acc[nt], false, false);
      acc[nt] = __builtin_amdgcn_wmma_f32_16x16x32_bf16(false, a1c.v, false, b1.v,
                                                        (short)0, acc[nt], false, false);
    }
    __syncthreads();                     // all waves done reading ldsB[buf]
  };

  // ---- prologue: step 0 prefetch ----
  issueB(0, 0, 0);
  V16U aX0, aX1, aY0, aY1;
  loadA(0, 0, aX0, aX1);

  // ---- 2x-unrolled ping-pong main loop (no register copies) ----
  for (int s = 0; s < S; s += 2) {
    step(s, 0, aX0, aX1, aY0, aY1);
    if (s + 1 < S) step(s + 1, 1, aY0, aY1, aX0, aX1);
  }

  #pragma unroll
  for (int nt = 0; nt < 4; ++nt) {
    int colOut = colBase + nt * 16 + m;
    #pragma unroll
    for (int d = 0; d < 8; ++d)
      out[(size_t)(tileM * 128 + wv * 16 + d + 8 * h) * N + colOut] = acc[nt][d];
  }
}

// ---------------------------------------------------------------------------
// Kernel 8/9: global mean pool + FC head
// ---------------------------------------------------------------------------
__global__ void meanpool_kernel(const float* __restrict__ x, float* __restrict__ out) {
  int t = blockIdx.x * blockDim.x + threadIdx.x;   // 8*512
  if (t >= 8 * 512) return;
  int bsm = t / 512, c = t - bsm * 512;
  float s = 0.f;
  for (int p = 0; p < 256; ++p) s += x[((size_t)bsm * 256 + p) * 512 + c];
  out[t] = s * (1.f / 256.f);
}

__global__ void fc_kernel(const float* __restrict__ y, const float* __restrict__ W,
                          const float* __restrict__ bias, float* __restrict__ out) {
  int t = blockIdx.x * blockDim.x + threadIdx.x;   // 8*1000
  if (t >= 8 * 1000) return;
  int bb = t / 1000, c = t - bb * 1000;
  float s = bias[c];
  for (int k = 0; k < 512; ++k) s = fmaf(y[bb * 512 + k], W[k * 1000 + c], s);
  out[t] = s;
}

// ---------------------------------------------------------------------------
// Host orchestration
// ---------------------------------------------------------------------------
static inline void launch_gemm(const unsigned short* A, const unsigned short* Wt, float* out,
                               int M, int N, int Cin, int K3, hipStream_t s) {
  int blocks = (M >> 7) * (N >> 6);                // 128x64 tile per 256-thread block
  wmma_gemm_kernel<<<blocks, 256, 0, s>>>(A, Wt, out, M, N, Cin, K3);
}

static inline void launch_conv(const float* x, const int* trip, int n_out, int Cin, int Cout,
                               const unsigned short* Wt, unsigned short* Abuf, float* out,
                               hipStream_t s) {
  scatter_kernel<<<n_out, 256, (size_t)(27 * Cin * 4), s>>>(x, trip, n_out, Cin, Abuf);
  launch_gemm(Abuf, Wt, out, n_out, Cout, Cin, 27, s);
}

extern "C" void kernel_launch(void* const* d_in, const int* in_sizes, int n_in,
                              void* d_out, int out_size, void* d_ws, size_t ws_size,
                              hipStream_t stream) {
  (void)in_sizes; (void)n_in; (void)out_size; (void)ws_size;
  // ---- input mapping (setup_inputs dict insertion order) ----
  const float* x0     = (const float*)d_in[0];
  const float* stem_w = (const float*)d_in[1];
  const float* stem_g = (const float*)d_in[2];
  const float* stem_b = (const float*)d_in[3];
  const float* fc_w   = (const float*)d_in[61];
  const float* fc_b   = (const float*)d_in[62];
  const int* trip_stem = (const int*)d_in[63];
  const int* pool_idx  = (const int*)d_in[64];
  const int* tS[4] = {(const int*)d_in[65], (const int*)d_in[66], (const int*)d_in[69], (const int*)d_in[72]};
  const int* tR[4] = {(const int*)d_in[65], (const int*)d_in[67], (const int*)d_in[70], (const int*)d_in[73]};
  const int* dsI[4] = {nullptr, (const int*)d_in[68], (const int*)d_in[71], (const int*)d_in[74]};
  const int P[4][2][9] = {
    {{4,5,6,7,8,9,-1,-1,-1},      {10,11,12,13,14,15,-1,-1,-1}},
    {{16,17,18,19,20,21,22,23,24},{25,26,27,28,29,30,-1,-1,-1}},
    {{31,32,33,34,35,36,37,38,39},{40,41,42,43,44,45,-1,-1,-1}},
    {{46,47,48,49,50,51,52,53,54},{55,56,57,58,59,60,-1,-1,-1}},
  };
  const int ch[4]   = {64, 128, 256, 512};
  const int inCh[4] = {64, 64, 128, 256};
  const int nL[4]   = {16384, 8192, 4096, 2048};   // B*N[l+1]

  // ---- workspace carve (all 256B aligned) ----
  char* p = (char*)d_ws;
  float* actA = (float*)p;  p += (size_t)8 << 20;
  float* actB = (float*)p;  p += (size_t)8 << 20;
  float* actC = (float*)p;  p += (size_t)8 << 20;
  float* actD = (float*)p;  p += (size_t)4 << 20;
  unsigned short* Abuf  = (unsigned short*)p; p += (size_t)58 << 20;
  unsigned short* dsbuf = (unsigned short*)p; p += (size_t)2 << 20;
  float* pooled = (float*)p; p += (size_t)1 << 16;
  unsigned short* WtBase = (unsigned short*)p;

  // ---- transpose + bf16-convert all conv/ds weights ----
  size_t wtOff[4][2][3];
  {
    size_t off = 0;
    for (int l = 0; l < 4; ++l)
      for (int blk = 0; blk < 2; ++blk) {
        int cin1 = (blk == 0) ? inCh[l] : ch[l];
        wtOff[l][blk][0] = off; off += (size_t)27 * cin1 * ch[l];
        wtOff[l][blk][1] = off; off += (size_t)27 * ch[l] * ch[l];
        if (blk == 0 && l > 0) { wtOff[l][blk][2] = off; off += (size_t)inCh[l] * ch[l]; }
      }
    for (int l = 0; l < 4; ++l)
      for (int blk = 0; blk < 2; ++blk) {
        int cin1 = (blk == 0) ? inCh[l] : ch[l];
        int tot1 = 27 * cin1 * ch[l];
        wt_kernel<<<(tot1 + 255) / 256, 256, 0, stream>>>(
            (const float*)d_in[P[l][blk][0]], WtBase + wtOff[l][blk][0], cin1, ch[l], tot1);
        int tot2 = 27 * ch[l] * ch[l];
        wt_kernel<<<(tot2 + 255) / 256, 256, 0, stream>>>(
            (const float*)d_in[P[l][blk][1]], WtBase + wtOff[l][blk][1], ch[l], ch[l], tot2);
        if (blk == 0 && l > 0) {
          int totd = inCh[l] * ch[l];
          wt_kernel<<<(totd + 255) / 256, 256, 0, stream>>>(
              (const float*)d_in[P[l][blk][6]], WtBase + wtOff[l][blk][2], inCh[l], ch[l], totd);
        }
      }
  }

  // ---- stem: conv(stride2) -> rln+relu -> maxpool ----
  stem_kernel<<<(32768 * 64) / 256, 256, 0, stream>>>(x0, trip_stem, stem_w, actB);
  rln_kernel<<<8, 1024, 0, stream>>>(actB, stem_g, stem_b, nullptr, actB, 4096, 64, 1);
  pool_kernel<<<(16384 * 64 + 255) / 256, 256, 0, stream>>>(actB, pool_idx, 16384, 64, actA);

  // ---- residual levels ----
  for (int l = 0; l < 4; ++l) {
    const int n_out = nL[l], C = ch[l], Cin0 = inCh[l], n_per = n_out / 8;
    // ----- block 0 -----
    launch_conv(actA, tS[l], n_out, Cin0, C, WtBase + wtOff[l][0][0], Abuf, actB, stream);
    rln_kernel<<<8, 1024, 0, stream>>>(actB, (const float*)d_in[P[l][0][2]], (const float*)d_in[P[l][0][3]],
                                       nullptr, actB, n_per, C, 1);
    launch_conv(actB, tR[l], n_out, C, C, WtBase + wtOff[l][0][1], Abuf, actC, stream);
    const float* resid;
    if (l == 0) {
      resid = actA;
    } else {
      gather_bf16_kernel<<<(n_out * Cin0 + 255) / 256, 256, 0, stream>>>(actA, dsI[l], n_out, Cin0, dsbuf);
      launch_gemm(dsbuf, WtBase + wtOff[l][0][2], actD, n_out, C, Cin0, 1, stream);
      rln_kernel<<<8, 1024, 0, stream>>>(actD, (const float*)d_in[P[l][0][7]], (const float*)d_in[P[l][0][8]],
                                         nullptr, actD, n_per, C, 0);
      resid = actD;
    }
    rln_kernel<<<8, 1024, 0, stream>>>(actC, (const float*)d_in[P[l][0][4]], (const float*)d_in[P[l][0][5]],
                                       resid, actA, n_per, C, 1);
    // ----- block 1 -----
    launch_conv(actA, tR[l], n_out, C, C, WtBase + wtOff[l][1][0], Abuf, actB, stream);
    rln_kernel<<<8, 1024, 0, stream>>>(actB, (const float*)d_in[P[l][1][2]], (const float*)d_in[P[l][1][3]],
                                       nullptr, actB, n_per, C, 1);
    launch_conv(actB, tR[l], n_out, C, C, WtBase + wtOff[l][1][1], Abuf, actC, stream);
    rln_kernel<<<8, 1024, 0, stream>>>(actC, (const float*)d_in[P[l][1][4]], (const float*)d_in[P[l][1][5]],
                                       actA, actA, n_per, C, 1);
  }

  // ---- head: ragged mean pool + FC ----
  meanpool_kernel<<<(8 * 512 + 255) / 256, 256, 0, stream>>>(actA, pooled);
  fc_kernel<<<(8 * 1000 + 255) / 256, 256, 0, stream>>>(pooled, fc_w, fc_b, (float*)d_out);
}
